// ShiftedWindowAttention_67199058313370
// MI455X (gfx1250) — matrix operand order repaired and checked by reading, not compiled, not run on hardware
//
#include <hip/hip_runtime.h>

typedef __bf16 v16bf __attribute__((ext_vector_type(16)));
typedef __bf16 v8bf  __attribute__((ext_vector_type(8)));
typedef float  v8f   __attribute__((ext_vector_type(8)));

#define NHEADS 12
#define HD     32
#define CDIM   384
#define QKVN   1152
#define NTOK   49
#define MPAD   64
#define WSZ    7
#define IMG    56
#define SHIFT  3

#define WQ_ELEMS (QKVN * CDIM)          // 442368
#define WP_ELEMS (CDIM * CDIM)          // 147456
#define RPB_ELEMS (NHEADS * NTOK * NTOK) // 28812

__device__ __forceinline__ v16bf join16(v8bf lo, v8bf hi) {
  union { v16bf v; v8bf h[2]; } u;
  u.h[0] = lo; u.h[1] = hi;
  return u.v;
}

// B-fragment from fp32 row-major [N,K] weight: 16 contiguous fp32 -> bf16
__device__ __forceinline__ v16bf ld_w_frag_f32(const float* __restrict__ w,
                                               int nrow, int ldw, int k0) {
  const float4* p = (const float4*)(w + nrow * ldw + k0);
  v16bf b;
#pragma unroll
  for (int j = 0; j < 4; ++j) {
    float4 f = p[j];
    b[4*j+0] = (__bf16)f.x; b[4*j+1] = (__bf16)f.y;
    b[4*j+2] = (__bf16)f.z; b[4*j+3] = (__bf16)f.w;
  }
  return b;
}

template <bool PRE>
__device__ __forceinline__ v16bf ld_w_frag(const void* __restrict__ w,
                                           int nrow, int ldw, int k0) {
  if (PRE) {
    return *(const v16bf*)((const __bf16*)w + (size_t)nrow * ldw + k0);
  } else {
    return ld_w_frag_f32((const float*)w, nrow, ldw, k0);
  }
}

__device__ __forceinline__ int region_id(int gh, int gw) {
  int rh = (gh < IMG - WSZ) ? 0 : ((gh < IMG - SHIFT) ? 1 : 2);
  int rw = (gw < IMG - WSZ) ? 0 : ((gw < IMG - SHIFT) ? 1 : 2);
  return rh * 3 + rw;
}

// ---- preprocessing: weights fp32->bf16, expand relative-position bias ----
__global__ __launch_bounds__(256) void swin_preprocess(
    const float* __restrict__ qkv_w, const float* __restrict__ proj_w,
    const float* __restrict__ rpb_table,
    __bf16* __restrict__ wq, __bf16* __restrict__ wp,
    float* __restrict__ rpb_full)
{
  const int nw4 = (WQ_ELEMS + WP_ELEMS) / 4;
  int i = blockIdx.x * 256 + threadIdx.x;
  if (i < nw4) {
    int idx4 = i * 4;
    const float* src;
    __bf16* dst;
    if (idx4 < WQ_ELEMS) { src = qkv_w + idx4; dst = wq + idx4; }
    else { src = proj_w + (idx4 - WQ_ELEMS); dst = wp + (idx4 - WQ_ELEMS); }
    const float4 f = *(const float4*)src;
    union { __bf16 h[4]; uint2 u; } pk;
    pk.h[0] = (__bf16)f.x; pk.h[1] = (__bf16)f.y;
    pk.h[2] = (__bf16)f.z; pk.h[3] = (__bf16)f.w;
    *(uint2*)dst = pk.u;
  } else if (i < nw4 + RPB_ELEMS) {
    int j = i - nw4;
    int h = j / (NTOK * NTOK);
    int rest = j % (NTOK * NTOK);
    int q = rest / NTOK, k = rest % NTOK;
    int qh = q / WSZ, qw = q % WSZ, kh = k / WSZ, kw = k % WSZ;
    rpb_full[j] = rpb_table[((qh - kh + 6) * 13 + (qw - kw + 6)) * NHEADS + h];
  }
}

template <bool PRE>
__global__ __launch_bounds__(256) void swin_attn_fused(
    const float* __restrict__ x,
    const void* __restrict__ qkv_w,    // bf16 if PRE else fp32, [1152,384]
    const void* __restrict__ proj_w,   // bf16 if PRE else fp32, [384,384]
    const float* __restrict__ qkv_b,
    const float* __restrict__ proj_b,
    const float* __restrict__ rpb,     // PRE: [12,49,49] expanded; else [169,12] table
    float* __restrict__ out)
{
  __shared__ __align__(64) __bf16 sX[MPAD * CDIM];      // x tile; reused as attn-out
  __shared__ __align__(64) __bf16 sQKV[MPAD * QKVN];    // qkv tile
  __shared__ __align__(64) __bf16 sP[8][16 * MPAD];     // per-wave softmax tile

  const int tid  = threadIdx.x;
  const int lane = tid & 31;
  const int wave = tid >> 5;
  const int nlo  = lane & 15;
  const int hsel = lane >> 4;
  const int akb  = hsel * 16;
  const int bkb  = hsel * 16;

  const int wid  = blockIdx.x;
  const int bidx = wid >> 6;
  const int win  = wid & 63;
  const int wy   = win >> 3;
  const int wx   = win & 7;

  // ---------- Phase 1: shifted window load, fp32 -> bf16 ----------
  for (int i = tid; i < MPAD * (CDIM / 4); i += 256) {
    int row = i / (CDIM / 4);
    int c4  = (i % (CDIM / 4)) * 4;
    union { __bf16 h[4]; uint2 u; } pk;
    if (row < NTOK) {
      int ph = row / WSZ, pw = row % WSZ;
      int sh = wy * WSZ + ph + SHIFT; if (sh >= IMG) sh -= IMG;
      int sw = wx * WSZ + pw + SHIFT; if (sw >= IMG) sw -= IMG;
      const float4 f = *(const float4*)(x + (((size_t)bidx * IMG + sh) * IMG + sw) * CDIM + c4);
      pk.h[0] = (__bf16)f.x; pk.h[1] = (__bf16)f.y;
      pk.h[2] = (__bf16)f.z; pk.h[3] = (__bf16)f.w;
    } else {
      pk.h[0] = pk.h[1] = pk.h[2] = pk.h[3] = (__bf16)0.0f;
    }
    *(uint2*)&sX[row * CDIM + c4] = pk.u;
  }
  __syncthreads();

  // ---------- Phase 2: QKV GEMM 64x1152x384, B-fragment reused across 4 M-tiles ----------
  for (int nt = wave; nt < QKVN / 16; nt += 8) {
    v8f acc[4] = {};
    int nrow = nt * 16 + nlo;
#pragma unroll 4
    for (int kk = 0; kk < CDIM / 32; ++kk) {
      int k0 = kk * 32;
      v16bf b = ld_w_frag<PRE>(qkv_w, nrow, CDIM, k0 + bkb);
#pragma unroll
      for (int mt = 0; mt < 4; ++mt) {
        int m = mt * 16 + nlo;
        v8bf alo = *(const v8bf*)&sX[m * CDIM + k0 + akb];
        v8bf ahi = *(const v8bf*)&sX[m * CDIM + k0 + 8 + akb];
        acc[mt] = __builtin_amdgcn_wmma_f32_16x16x32_bf16(
            false, join16(alo, ahi), false, b, (short)0, acc[mt], false, false);
      }
    }
    int col = nt * 16 + nlo;
    float bias = qkv_b[col];
#pragma unroll
    for (int mt = 0; mt < 4; ++mt)
#pragma unroll
      for (int r = 0; r < 8; ++r) {
        int row = mt * 16 + r + hsel * 8;
        sQKV[row * QKVN + col] = (__bf16)(acc[mt][r] + bias);
      }
  }
  __syncthreads();

  // ---------- Phase 3: attention per head ----------
  const float scale = 0.17677669529663687f; // 1/sqrt(32)
  for (int h = wave; h < NHEADS; h += 8) {
    v16bf kf[4];
#pragma unroll
    for (int nt = 0; nt < 4; ++nt) {
      int tok = nt * 16 + nlo;
      kf[nt] = *(const v16bf*)&sQKV[tok * QKVN + CDIM + h * HD + bkb];
    }
    v16bf vf[2][2];
#pragma unroll
    for (int ks = 0; ks < 2; ++ks)
#pragma unroll
      for (int ot = 0; ot < 2; ++ot) {
        int kb0  = ks * 32 + hsel * 16;
        int ncol = 2 * CDIM + h * HD + ot * 16 + nlo;
        v16bf v;
#pragma unroll
        for (int i = 0; i < 16; ++i)
          v[i] = sQKV[(kb0 + i) * QKVN + ncol];
        vf[ks][ot] = v;
      }

    for (int mt = 0; mt < 4; ++mt) {
      int mrow = mt * 16 + nlo;
      v8bf qlo = *(const v8bf*)&sQKV[mrow * QKVN + h * HD + akb];
      v8bf qhi = *(const v8bf*)&sQKV[mrow * QKVN + h * HD + 8 + akb];
      v16bf qf = join16(qlo, qhi);
      v8f s[4];
#pragma unroll
      for (int nt = 0; nt < 4; ++nt) {
        v8f z = {};
        s[nt] = __builtin_amdgcn_wmma_f32_16x16x32_bf16(
            false, qf, false, kf[nt], (short)0, z, false, false);
      }
#pragma unroll
      for (int r = 0; r < 8; ++r) {
        int row = mt * 16 + r + hsel * 8;
        bool rowok = row < NTOK;
        int qh = rowok ? row / WSZ : 0, qw = rowok ? row % WSZ : 0;
        int rq = region_id(wy * WSZ + qh, wx * WSZ + qw);
        float vals[4];
#pragma unroll
        for (int nt = 0; nt < 4; ++nt) {
          int col = nt * 16 + nlo;
          float v = s[nt][r] * scale;
          if (col < NTOK && rowok) {
            if (PRE)
              v += rpb[(h * NTOK + row) * NTOK + col];
            else {
              int kh = col / WSZ, kw = col % WSZ;
              v += rpb[((qh - kh + 6) * 13 + (qw - kw + 6)) * NHEADS + h];
            }
            int kh = col / WSZ, kw = col % WSZ;
            int rk = region_id(wy * WSZ + kh, wx * WSZ + kw);
            if (rk != rq) v -= 100.0f;
          }
          if (col >= NTOK) v = -1e30f;
          vals[nt] = v;
        }
        float mx = fmaxf(fmaxf(vals[0], vals[1]), fmaxf(vals[2], vals[3]));
#pragma unroll
        for (int sm = 1; sm < 16; sm <<= 1)
          mx = fmaxf(mx, __shfl_xor(mx, sm, 32));
        float sum = 0.f;
#pragma unroll
        for (int nt = 0; nt < 4; ++nt) {
          float e = __expf(vals[nt] - mx);
          vals[nt] = e;
          sum += e;
        }
#pragma unroll
        for (int sm = 1; sm < 16; sm <<= 1)
          sum += __shfl_xor(sum, sm, 32);
        float inv = 1.0f / sum;
#pragma unroll
        for (int nt = 0; nt < 4; ++nt)
          sP[wave][(r + hsel * 8) * MPAD + nt * 16 + nlo] = (__bf16)(vals[nt] * inv);
      }
      // O = P * V  (16x32 out, K=64 in 2 steps)
      v8f o0 = {}, o1 = {};
#pragma unroll
      for (int ks = 0; ks < 2; ++ks) {
        int k0 = ks * 32;
        v8bf plo = *(const v8bf*)&sP[wave][nlo * MPAD + k0 + akb];
        v8bf phi = *(const v8bf*)&sP[wave][nlo * MPAD + k0 + 8 + akb];
        v16bf pf = join16(plo, phi);
        o0 = __builtin_amdgcn_wmma_f32_16x16x32_bf16(false, pf, false, vf[ks][0], (short)0, o0, false, false);
        o1 = __builtin_amdgcn_wmma_f32_16x16x32_bf16(false, pf, false, vf[ks][1], (short)0, o1, false, false);
      }
#pragma unroll
      for (int r = 0; r < 8; ++r) {
        int row = mt * 16 + r + hsel * 8;
        sX[row * CDIM + h * HD + nlo]      = (__bf16)o0[r];
        sX[row * CDIM + h * HD + 16 + nlo] = (__bf16)o1[r];
      }
    }
  }
  __syncthreads();

  // ---------- Phase 4: proj GEMM 64x384x384 + window-reverse store ----------
  for (int nt = wave; nt < CDIM / 16; nt += 8) {
    v8f acc[4] = {};
    int nrow = nt * 16 + nlo;
#pragma unroll 4
    for (int kk = 0; kk < CDIM / 32; ++kk) {
      int k0 = kk * 32;
      v16bf b = ld_w_frag<PRE>(proj_w, nrow, CDIM, k0 + bkb);
#pragma unroll
      for (int mt = 0; mt < 4; ++mt) {
        int m = mt * 16 + nlo;
        v8bf alo = *(const v8bf*)&sX[m * CDIM + k0 + akb];
        v8bf ahi = *(const v8bf*)&sX[m * CDIM + k0 + 8 + akb];
        acc[mt] = __builtin_amdgcn_wmma_f32_16x16x32_bf16(
            false, join16(alo, ahi), false, b, (short)0, acc[mt], false, false);
      }
    }
    int col = nt * 16 + nlo;
    float bias = proj_b[col];
#pragma unroll
    for (int mt = 0; mt < 4; ++mt)
#pragma unroll
      for (int r = 0; r < 8; ++r) {
        int row = mt * 16 + r + hsel * 8;
        if (row < NTOK) {
          int ph = row / WSZ, pw = row % WSZ;
          int oh = wy * WSZ + ph + SHIFT; if (oh >= IMG) oh -= IMG;
          int ow = wx * WSZ + pw + SHIFT; if (ow >= IMG) ow -= IMG;
          out[(((size_t)bidx * IMG + oh) * IMG + ow) * CDIM + col] = acc[mt][r] + bias;
        }
      }
  }
}

extern "C" void kernel_launch(void* const* d_in, const int* in_sizes, int n_in,
                              void* d_out, int out_size, void* d_ws, size_t ws_size,
                              hipStream_t stream) {
  const float* x      = (const float*)d_in[0];
  const float* qkv_w  = (const float*)d_in[1];
  const float* qkv_b  = (const float*)d_in[2];
  const float* proj_w = (const float*)d_in[3];
  const float* proj_b = (const float*)d_in[4];
  const float* rpb    = (const float*)d_in[5];
  float* outp = (float*)d_out;

  dim3 grid(32 * 64);
  dim3 block(256);

  const size_t need = (size_t)(WQ_ELEMS + WP_ELEMS) * sizeof(__bf16)
                    + (size_t)RPB_ELEMS * sizeof(float);
  if (d_ws != nullptr && ws_size >= need) {
    __bf16* wq = (__bf16*)d_ws;
    __bf16* wp = wq + WQ_ELEMS;
    float*  rpb_full = (float*)(wp + WP_ELEMS);
    const int pre_items = (WQ_ELEMS + WP_ELEMS) / 4 + RPB_ELEMS;
    hipLaunchKernelGGL(swin_preprocess, dim3((pre_items + 255) / 256), block, 0, stream,
                       qkv_w, proj_w, rpb, wq, wp, rpb_full);
    hipLaunchKernelGGL((swin_attn_fused<true>), grid, block, 0, stream,
                       x, (const void*)wq, (const void*)wp, qkv_b, proj_b, rpb_full, outp);
  } else {
    hipLaunchKernelGGL((swin_attn_fused<false>), grid, block, 0, stream,
                       x, (const void*)qkv_w, (const void*)proj_w, qkv_b, proj_b, rpb, outp);
  }
}
